// SlotAttention_65412351918168
// MI455X (gfx1250) — matrix operand order repaired and checked by reading, not compiled, not run on hardware
//
#include <hip/hip_runtime.h>

// ---------------------------------------------------------------------------
// CDNA5 (gfx1250, wave32) WMMA slot-attention pipeline, bf16 in / f32 accum.
// GEMMs register-tiled 16x64 per wave: 1 A-fragment feeds 4 WMMAs.
// ---------------------------------------------------------------------------

typedef __attribute__((ext_vector_type(16))) __bf16 v16bf;
typedef __attribute__((ext_vector_type(8)))  __bf16 v8bf;
typedef __attribute__((ext_vector_type(8)))  float  v8f;

union Frag32B { v16bf v; v8bf h[2]; };

// A-fragment (16x32 bf16, MxK): lane m = lane&15, half = lane>>4.
// elements 0..7  = A[m][k + half*8 .. +7]
// elements 8..15 = A[m][k + 16 + half*8 .. +7]
__device__ __forceinline__ v16bf ld_fragA(const __bf16* __restrict__ row, int k, int hf) {
  Frag32B f;
  f.h[0] = *(const v8bf*)(row + k + hf * 8);
  f.h[1] = *(const v8bf*)(row + k + 16 + hf * 8);
  return f.v;
}
// B-fragment (32x16 bf16, KxN) from B^T row-major: lane n = lane&15.
// elements 0..7  = BT[n][k + half*8 .. +7]
// elements 8..15 = BT[n][k + 16 + half*8 .. +7]
__device__ __forceinline__ v16bf ld_fragB(const __bf16* __restrict__ row, int k, int hf) {
  Frag32B f;
  f.h[0] = *(const v8bf*)(row + k + hf * 8);
  f.h[1] = *(const v8bf*)(row + k + 16 + hf * 8);
  return f.v;
}

__device__ __forceinline__ v8f wmma_bf16(v16bf a, v16bf b, v8f c) {
  return __builtin_amdgcn_wmma_f32_16x16x32_bf16(false, a, false, b, (short)0, c, false, false);
}

#define GF_OUTF32 1
#define GF_TRANS  2
#define GF_RELU   4

// Generic GEMM: C = alpha * (A @ BT^T) + bias + addend.
// A: [batch][M x K] bf16, BT: [batch][N x K] bf16 (i.e. B transposed).
// One wave computes a 16(M) x 64(N) tile: 4 accumulators, A-frag reused 4x.
// Requires M % 16 == 0, N % 64 == 0, K % 32 == 0. blockDim.x must be 128.
__global__ void gemm_wmma(const __bf16* __restrict__ A, long long sA, int lda,
                          const __bf16* __restrict__ BT, long long sB, int ldb,
                          const float* __restrict__ bias,
                          const float* __restrict__ addend, long long sAdd, int ldAdd,
                          float alpha, int M, int N, int K,
                          void* __restrict__ out, long long sC, int ldc, int flags) {
  const int lane = threadIdx.x & 31;
  const int wid  = (blockIdx.x << 2) + (threadIdx.x >> 5);
  const int nt   = N >> 6;                  // 64-wide column tiles
  const int tiles = (M >> 4) * nt;
  if (wid >= tiles) return;                 // wave-uniform exit (EXEC stays all-ones)
  const int tm = wid / nt, tn = wid - tm * nt;
  const int bb = blockIdx.y;
  const int idx = lane & 15, hf = lane >> 4;
  const __bf16* arow = A  + bb * sA + (long long)(tm * 16 + idx) * lda;
  const __bf16* brow = BT + bb * sB + (long long)(tn * 64 + idx) * ldb;
  const long long bstep = (long long)16 * ldb;
  v8f acc[4];
#pragma unroll
  for (int j = 0; j < 4; ++j) acc[j] = (v8f){0.f, 0.f, 0.f, 0.f, 0.f, 0.f, 0.f, 0.f};
  for (int k = 0; k < K; k += 32) {
    const v16bf af = ld_fragA(arow, k, hf);
    const v16bf b0 = ld_fragB(brow,             k, hf);
    const v16bf b1 = ld_fragB(brow +     bstep, k, hf);
    const v16bf b2 = ld_fragB(brow + 2 * bstep, k, hf);
    const v16bf b3 = ld_fragB(brow + 3 * bstep, k, hf);
    acc[0] = wmma_bf16(af, b0, acc[0]);
    acc[1] = wmma_bf16(af, b1, acc[1]);
    acc[2] = wmma_bf16(af, b2, acc[2]);
    acc[3] = wmma_bf16(af, b3, acc[3]);
  }
#pragma unroll
  for (int j = 0; j < 4; ++j) {
    const int col = tn * 64 + j * 16 + idx;
    const float bv = bias ? bias[col] : 0.0f;
#pragma unroll
    for (int r = 0; r < 8; ++r) {
      const int row = tm * 16 + r + hf * 8;   // C layout: VGPR r -> M = r + 8*half
      float val = acc[j][r] * alpha + bv;
      if (addend) val += addend[bb * sAdd + (long long)row * ldAdd + col];
      if (flags & GF_RELU) val = fmaxf(val, 0.0f);
      const long long off = (flags & GF_TRANS)
          ? (bb * sC + (long long)col * ldc + row)
          : (bb * sC + (long long)row * ldc + col);
      if (flags & GF_OUTF32) ((float*)out)[off] = val;
      else                   ((__bf16*)out)[off] = (__bf16)val;
    }
  }
}

// updates[b,s,d] = sum_n attn[b,s,n] * v[b,n,d]; v stored transposed vT[b,d,n].
// Split-K: 8 waves per block each take K-chunk of 512, LDS tree-reduce.
// grid = (16 d-tiles, 32 batches), blockDim = 256.
__global__ void gemm_updates_splitk(const __bf16* __restrict__ attn,
                                    const __bf16* __restrict__ vT,
                                    __bf16* __restrict__ upd) {
  __shared__ float red[8][16][16];
  const int lane = threadIdx.x & 31;
  const int w = threadIdx.x >> 5;
  const int dt = blockIdx.x;
  const int b  = blockIdx.y;
  const int idx = lane & 15, hf = lane >> 4;
  const __bf16* arow = attn + (long long)b * 16 * 4096 + (long long)idx * 4096;
  const __bf16* brow = vT   + (long long)b * 256 * 4096 + (long long)(dt * 16 + idx) * 4096;
  v8f acc = {0.f, 0.f, 0.f, 0.f, 0.f, 0.f, 0.f, 0.f};
  const int k0 = w * 512;
  for (int k = k0; k < k0 + 512; k += 32) {
    acc = wmma_bf16(ld_fragA(arow, k, hf), ld_fragB(brow, k, hf), acc);
  }
#pragma unroll
  for (int r = 0; r < 8; ++r) red[w][r + hf * 8][idx] = acc[r];
  __syncthreads();
  const int row = threadIdx.x >> 4;   // 0..15 (slot)
  const int col = threadIdx.x & 15;   // 0..15 (d within tile)
  float s = 0.f;
#pragma unroll
  for (int i = 0; i < 8; ++i) s += red[i][row][col];
  upd[(long long)(b * 16 + row) * 256 + dt * 16 + col] = (__bf16)s;
}

// LayerNorm over 256 columns, one wave per row, bf16 output. blockDim = 256.
__global__ void ln256_bf16(const float* __restrict__ in, const float* __restrict__ gam,
                           const float* __restrict__ bet, __bf16* __restrict__ out, int rows) {
  const int w = (blockIdx.x << 3) + (threadIdx.x >> 5);
  const int lane = threadIdx.x & 31;
  if (w >= rows) return;
  const float* row = in + (long long)w * 256;
  float x[8]; float s = 0.f;
#pragma unroll
  for (int i = 0; i < 8; ++i) { x[i] = row[lane + i * 32]; s += x[i]; }
#pragma unroll
  for (int o = 16; o > 0; o >>= 1) s += __shfl_xor(s, o, 32);
  const float mean = s * (1.0f / 256.0f);
  float vs = 0.f;
#pragma unroll
  for (int i = 0; i < 8; ++i) { float d = x[i] - mean; vs += d * d; }
#pragma unroll
  for (int o = 16; o > 0; o >>= 1) vs += __shfl_xor(vs, o, 32);
  const float rstd = rsqrtf(vs * (1.0f / 256.0f) + 1e-5f);
  __bf16* orow = out + (long long)w * 256;
#pragma unroll
  for (int i = 0; i < 8; ++i) {
    const int c = lane + i * 32;
    orow[c] = (__bf16)((x[i] - mean) * rstd * gam[c] + bet[c]);
  }
}

// Softmax over S=16 slots for each (b,n) column. Writes bf16 attn for WMMA and
// optionally fp32 attn (final iteration output).
__global__ void softmax16_k(const float* __restrict__ logits, __bf16* __restrict__ attn_bf,
                            float* __restrict__ attn_f32) {
  const int idx = blockIdx.x * 256 + threadIdx.x;
  if (idx >= 32 * 4096) return;
  const int b = idx >> 12;
  const int n = idx & 4095;
  const float* p = logits + b * 65536 + n;
  float v[16];
  float mx = -3.4e38f;
#pragma unroll
  for (int s = 0; s < 16; ++s) { v[s] = p[s * 4096]; mx = fmaxf(mx, v[s]); }
  float sum = 0.f;
#pragma unroll
  for (int s = 0; s < 16; ++s) { v[s] = expf(v[s] - mx); sum += v[s]; }
  const float inv = 1.0f / sum;
  __bf16* q = attn_bf + b * 65536 + n;
#pragma unroll
  for (int s = 0; s < 16; ++s) {
    const float a = v[s] * inv;
    q[s * 4096] = (__bf16)a;
    if (attn_f32) attn_f32[b * 65536 + s * 4096 + n] = a;
  }
}

// GRU gate math (PyTorch GRUCell), rows = B*S = 512, cols = 256.
__global__ void gru_gates_k(const float* __restrict__ gi, const float* __restrict__ gh,
                            const float* __restrict__ hprev, float* __restrict__ h) {
  const int idx = blockIdx.x * 256 + threadIdx.x;
  if (idx >= 512 * 256) return;
  const int row = idx >> 8, col = idx & 255;
  const int base = row * 768 + col;
  const float ir = gi[base], iz = gi[base + 256], inn = gi[base + 512];
  const float hr = gh[base], hz = gh[base + 256], hn = gh[base + 512];
  const float r = 1.0f / (1.0f + expf(-(ir + hr)));
  const float z = 1.0f / (1.0f + expf(-(iz + hz)));
  const float n = tanhf(inn + r * hn);
  h[idx] = (1.0f - z) * n + z * hprev[idx];
}

__global__ void init_slots_k(const float* __restrict__ mu, const float* __restrict__ sg,
                             const float* __restrict__ noise, float* __restrict__ slots) {
  const int idx = blockIdx.x * 256 + threadIdx.x;
  if (idx >= 512 * 256) return;
  const int d = idx & 255;
  slots[idx] = mu[d] + sg[d] * noise[idx];
}

__global__ void cvt_f32_to_bf16(const float* __restrict__ in, __bf16* __restrict__ out, int n) {
  const int i = blockIdx.x * 256 + threadIdx.x;
  if (i < n) out[i] = (__bf16)in[i];
}

// ---------------------------------------------------------------------------

extern "C" void kernel_launch(void* const* d_in, const int* in_sizes, int n_in,
                              void* d_out, int out_size, void* d_ws, size_t ws_size,
                              hipStream_t stream) {
  const float* inputs     = (const float*)d_in[0];
  const float* init_noise = (const float*)d_in[1];
  const float* slots_mu   = (const float*)d_in[2];
  const float* slots_sig  = (const float*)d_in[3];
  const float* ln_in_g = (const float*)d_in[4];
  const float* ln_in_b = (const float*)d_in[5];
  const float* ln_s_g  = (const float*)d_in[6];
  const float* ln_s_b  = (const float*)d_in[7];
  const float* ln_m_g  = (const float*)d_in[8];
  const float* ln_m_b  = (const float*)d_in[9];
  const float* Wk = (const float*)d_in[10]; const float* bk = (const float*)d_in[11];
  const float* Wv = (const float*)d_in[12]; const float* bv = (const float*)d_in[13];
  const float* Wq = (const float*)d_in[14]; const float* bq = (const float*)d_in[15];
  const float* W_ih = (const float*)d_in[16]; const float* W_hh = (const float*)d_in[17];
  const float* b_ih = (const float*)d_in[18]; const float* b_hh = (const float*)d_in[19];
  const float* W1 = (const float*)d_in[20]; const float* b1 = (const float*)d_in[21];
  const float* W2 = (const float*)d_in[22]; const float* b2 = (const float*)d_in[23];

  const int BN = 32 * 4096;     // 131072 rows of input
  const int BS = 32 * 16;       // 512 slot rows

  // ---- workspace bump allocator (256B aligned) ----
  char* wp = (char*)d_ws;
  auto alloc = [&](size_t bytes) -> void* {
    void* r = (void*)wp;
    wp += (bytes + 255) & ~(size_t)255;
    return r;
  };
  __bf16* wk_bf  = (__bf16*)alloc(256 * 256 * 2);
  __bf16* wv_bf  = (__bf16*)alloc(256 * 256 * 2);
  __bf16* wq_bf  = (__bf16*)alloc(256 * 256 * 2);
  __bf16* wih_bf = (__bf16*)alloc(768 * 256 * 2);
  __bf16* whh_bf = (__bf16*)alloc(768 * 256 * 2);
  __bf16* w1_bf  = (__bf16*)alloc(512 * 256 * 2);
  __bf16* w2_bf  = (__bf16*)alloc(256 * 512 * 2);
  __bf16* xln    = (__bf16*)alloc((size_t)BN * 256 * 2);   // LN(inputs), bf16
  __bf16* k_bf   = (__bf16*)alloc((size_t)BN * 256 * 2);   // [B,N,D]
  __bf16* vT_bf  = (__bf16*)alloc((size_t)BN * 256 * 2);   // [B,D,N]
  float*  slots    = (float*)alloc((size_t)BS * 256 * 4);
  __bf16* slots_bf = (__bf16*)alloc((size_t)BS * 256 * 2);
  __bf16* sn_bf    = (__bf16*)alloc((size_t)BS * 256 * 2);
  __bf16* q_bf     = (__bf16*)alloc((size_t)BS * 256 * 2);
  float*  logits   = (float*)alloc((size_t)32 * 16 * 4096 * 4);
  __bf16* attn_bf  = (__bf16*)alloc((size_t)32 * 16 * 4096 * 2);
  __bf16* upd_bf   = (__bf16*)alloc((size_t)BS * 256 * 2);
  float*  gi       = (float*)alloc((size_t)BS * 768 * 4);
  float*  gh       = (float*)alloc((size_t)BS * 768 * 4);
  float*  hbuf     = (float*)alloc((size_t)BS * 256 * 4);
  __bf16* m_bf     = (__bf16*)alloc((size_t)BS * 256 * 2);
  __bf16* t_bf     = (__bf16*)alloc((size_t)BS * 512 * 2);

  auto cvt = [&](const float* src, __bf16* dst, int n) {
    cvt_f32_to_bf16<<<(n + 255) / 256, 256, 0, stream>>>(src, dst, n);
  };
  auto gemm = [&](const __bf16* A, long long sA, int lda,
                  const __bf16* BT, long long sB, int ldb,
                  const float* bias, const float* addend, long long sAdd, int ldAdd,
                  float alpha, int M, int N, int K,
                  void* out, long long sC, int ldc, int flags, int batches) {
    const int waves = (M >> 4) * (N >> 6);   // 16x64 tile per wave
    dim3 grid((waves + 3) >> 2, batches);
    gemm_wmma<<<grid, 128, 0, stream>>>(A, sA, lda, BT, sB, ldb, bias,
                                        addend, sAdd, ldAdd, alpha, M, N, K,
                                        out, sC, ldc, flags);
  };

  // ---- one-time per call: weights -> bf16, slots init, LN(inputs), K/V proj
  cvt(Wk,   wk_bf,  256 * 256);
  cvt(Wv,   wv_bf,  256 * 256);
  cvt(Wq,   wq_bf,  256 * 256);
  cvt(W_ih, wih_bf, 768 * 256);
  cvt(W_hh, whh_bf, 768 * 256);
  cvt(W1,   w1_bf,  512 * 256);
  cvt(W2,   w2_bf,  256 * 512);

  init_slots_k<<<(BS * 256 + 255) / 256, 256, 0, stream>>>(slots_mu, slots_sig, init_noise, slots);
  ln256_bf16<<<(BN + 7) / 8, 256, 0, stream>>>(inputs, ln_in_g, ln_in_b, xln, BN);

  // k = xln @ Wk^T + bk  -> bf16 [B*N, 256]
  gemm(xln, 0, 256, wk_bf, 0, 256, bk, nullptr, 0, 0,
       1.0f, BN, 256, 256, k_bf, 0, 256, /*bf16*/0, 1);
  // v^T = (xln @ Wv^T + bv)^T -> bf16 [B, 256, 4096] (batched, transposed store)
  gemm(xln, (long long)4096 * 256, 256, wv_bf, 0, 256, bv, nullptr, 0, 0,
       1.0f, 4096, 256, 256, vT_bf, (long long)256 * 4096, 4096, GF_TRANS, 32);

  float* outF = (float*)d_out;             // [0,131072): slots ; [131072,...): attn

  for (int it = 0; it < 3; ++it) {
    // sn = LN(slots); slots_prev -> bf16
    ln256_bf16<<<(BS + 7) / 8, 256, 0, stream>>>(slots, ln_s_g, ln_s_b, sn_bf, BS);
    cvt(slots, slots_bf, BS * 256);
    // q = sn @ Wq^T + bq -> bf16 [B,16,256]
    gemm(sn_bf, 0, 256, wq_bf, 0, 256, bq, nullptr, 0, 0,
         1.0f, BS, 256, 256, q_bf, 0, 256, 0, 1);
    // logits = (q @ k^T) * D^-0.5 -> f32 [B,16,4096] (batched)
    gemm(q_bf, 16 * 256, 256, k_bf, (long long)4096 * 256, 256, nullptr, nullptr, 0, 0,
         0.0625f, 16, 4096, 256, logits, 16 * 4096, 4096, GF_OUTF32, 32);
    // softmax over slots; final iteration also writes fp32 attn to d_out
    softmax16_k<<<(BN + 255) / 256, 256, 0, stream>>>(
        logits, attn_bf, (it == 2) ? (outF + 131072) : nullptr);
    // updates = attn @ v -> bf16 [B*16, 256] (split-K over N=4096)
    gemm_updates_splitk<<<dim3(16, 32), 256, 0, stream>>>(attn_bf, vT_bf, upd_bf);
    // GRU gate pre-activations
    gemm(upd_bf,   0, 256, wih_bf, 0, 256, b_ih, nullptr, 0, 0,
         1.0f, BS, 768, 256, gi, 0, 768, GF_OUTF32, 1);
    gemm(slots_bf, 0, 256, whh_bf, 0, 256, b_hh, nullptr, 0, 0,
         1.0f, BS, 768, 256, gh, 0, 768, GF_OUTF32, 1);
    gru_gates_k<<<(BS * 256 + 255) / 256, 256, 0, stream>>>(gi, gh, slots, hbuf);
    // m = LN(h) -> bf16; MLP with ReLU; residual add of h
    ln256_bf16<<<(BS + 7) / 8, 256, 0, stream>>>(hbuf, ln_m_g, ln_m_b, m_bf, BS);
    gemm(m_bf, 0, 256, w1_bf, 0, 256, b1, nullptr, 0, 0,
         1.0f, BS, 512, 256, t_bf, 0, 512, GF_RELU, 1);
    float* sdst = (it == 2) ? outF : slots;
    gemm(t_bf, 0, 512, w2_bf, 0, 512, b2, hbuf, 0, 256,
         1.0f, BS, 256, 512, sdst, 0, 256, GF_OUTF32, 1);
  }
}
